// HighDimKernelRegressor_89601607729551
// MI455X (gfx1250) — matrix operand order repaired and checked by reading, not compiled, not run on hardware
//
#include <hip/hip_runtime.h>
#include <hip/hip_bf16.h>

typedef __attribute__((ext_vector_type(16))) _Float16 v16h;
typedef __attribute__((ext_vector_type(8)))  float    v8f;

#define N_NODES 2048
#define H_DIM   64
#define LEAKY   0.2f

union HV { v16h v; _Float16 h[16]; unsigned u[8]; };

__device__ __forceinline__ float fast_sigmoid(float x) {
    // 1/(1+exp(-x)) with v_exp_f32 + v_rcp_f32 (avoids IEEE div sequence)
    return __builtin_amdgcn_rcpf(1.0f + __expf(-x));
}

// Transposed mapping: D' = W1 x diff^T per 16-pair tile.
//   A (static, per wave) = W1 rows in A-layout: A[nh][kb], M = n-half row,
//     K = d: VGPR v -> d = kb*32 + 16*hi16 + 2v (+1)  (contiguous, as B)
//   B (per i-step)      = f16(|h_j - h_i|) in B-layout: B[kb], N = m = l15,
//     K = d: VGPR v -> d = kb*32 + 16*hi16 + 2v (+1)  (contiguous chunks)
//   D' C-layout: VGPR r, lane l -> n = nh*16 + 8*hi16 + r, m = l15
// fc2 is a per-lane reduction over 16 n's (4 independent accumulators to
// break the fmac dependency chain) + one shfl_xor(16); result for pair m
// lives in lane m -> coalesced 64B row store.
__global__ __launch_bounds__(256) void simnet_allpairs_wmma(
    const float* __restrict__ hiddens,
    const float* __restrict__ W1,
    const float* __restrict__ b1,
    const float* __restrict__ W2,
    const float* __restrict__ b2,
    float* __restrict__ outKernel)
{
    __shared__ float lds_hi[16 * H_DIM];

    const int tid  = threadIdx.x;
    const int lane = tid & 31;
    const int wave = tid >> 5;
    const int l15  = lane & 15;
    const int hi16 = lane >> 4;   // 0: lanes 0-15, 1: lanes 16-31

    const int ibase = blockIdx.y * 16;
    const int jbase = blockIdx.x * 128 + wave * 16;

    // ---- cooperative LDS fill of the i-tile: 16x64 f32 = 256 float4 ----
    {
        const float4* src = (const float4*)(hiddens + (size_t)ibase * H_DIM);
        float4*       dst = (float4*)lds_hi;
        dst[tid] = src[tid];
    }
    __syncthreads();

    // ---- per-lane j-row data: d in [kb*32 + 16*hi16, +16), kb = 0,1 ----
    const int j = jbase + l15;
    const float* hrow = hiddens + (size_t)j * H_DIM;
    __builtin_prefetch(hrow, 0, 1);
    float hj2[2][16];
#pragma unroll
    for (int kb = 0; kb < 2; ++kb) {
        const float4* p = (const float4*)(hrow + kb * 32 + 16 * hi16);
#pragma unroll
        for (int q = 0; q < 4; ++q) {
            float4 x = p[q];
            hj2[kb][4 * q + 0] = x.x; hj2[kb][4 * q + 1] = x.y;
            hj2[kb][4 * q + 2] = x.z; hj2[kb][4 * q + 3] = x.w;
        }
    }

    // ---- static A operands: W1 rows in f16 A-layout ----
    HV Am[2][2];
#pragma unroll
    for (int nh = 0; nh < 2; ++nh) {
        const float* wrow = W1 + (size_t)(nh * 16 + l15) * H_DIM;
#pragma unroll
        for (int kb = 0; kb < 2; ++kb) {
#pragma unroll
            for (int v = 0; v < 8; ++v) {
                const int d = kb * 32 + 16 * hi16 + 2 * v;
                auto pk = __builtin_amdgcn_cvt_pkrtz(wrow[d], wrow[d + 1]);
                Am[nh][kb].u[v] = __builtin_bit_cast(unsigned, pk);
            }
        }
    }

    // ---- per-lane fc2 weights/biases: n = g*16 + 8*hi16 + r ----
    float w2v[16], b1v[16];
#pragma unroll
    for (int g = 0; g < 2; ++g) {
        const float* wp = W2 + g * 16 + 8 * hi16;
        const float* bp = b1 + g * 16 + 8 * hi16;
#pragma unroll
        for (int r = 0; r < 8; ++r) {
            w2v[8 * g + r] = wp[r];
            b1v[8 * g + r] = bp[r];
        }
    }
    const float bias2 = b2[0];

    // ---- loop over the 16 i rows of this block ----
    for (int ii = 0; ii < 16; ++ii) {
        // B = f16(|h_j - h_i|): consecutive d-pairs, abs via sign-bit clear
        // after v_cvt_pk_rtz (f32 fabs on v2f32 crashes gfx1250 ISel).
        HV Bd[2];
#pragma unroll
        for (int kb = 0; kb < 2; ++kb) {
            const float4* hp = (const float4*)(lds_hi + ii * H_DIM + kb * 32 + 16 * hi16);
            float4 y0 = hp[0], y1 = hp[1], y2 = hp[2], y3 = hp[3];
            float hic[16] = { y0.x, y0.y, y0.z, y0.w, y1.x, y1.y, y1.z, y1.w,
                              y2.x, y2.y, y2.z, y2.w, y3.x, y3.y, y3.z, y3.w };
#pragma unroll
            for (int v = 0; v < 8; ++v) {
                float d0 = hj2[kb][2 * v]     - hic[2 * v];
                float d1 = hj2[kb][2 * v + 1] - hic[2 * v + 1];
                auto pk = __builtin_amdgcn_cvt_pkrtz(d0, d1);
                Bd[kb].u[v] = __builtin_bit_cast(unsigned, pk) & 0x7FFF7FFFu;
            }
        }

        // fc1 (transposed): D'[nh] = W1[nh] x diff^T, accumulate over K-blocks
        v8f d0 = {};  // n = 0..15  (per lane: n = 8*hi16 + r)
        v8f d1 = {};  // n = 16..31 (per lane: n = 16 + 8*hi16 + r)
        d0 = __builtin_amdgcn_wmma_f32_16x16x32_f16(false, Am[0][0].v, false, Bd[0].v,
                                                    (short)0, d0, false, false);
        d0 = __builtin_amdgcn_wmma_f32_16x16x32_f16(false, Am[0][1].v, false, Bd[1].v,
                                                    (short)0, d0, false, false);
        d1 = __builtin_amdgcn_wmma_f32_16x16x32_f16(false, Am[1][0].v, false, Bd[0].v,
                                                    (short)0, d1, false, false);
        d1 = __builtin_amdgcn_wmma_f32_16x16x32_f16(false, Am[1][1].v, false, Bd[1].v,
                                                    (short)0, d1, false, false);

        // fc2: per-lane reduction over this lane's 16 n's, 4 independent
        // accumulators to break the fmac dependency chain.
        // leaky(x) = max(x, 0.2x) since 0 < slope < 1.
        float a0 = 0.0f, a1 = 0.0f, a2 = 0.0f, a3 = 0.0f;
#pragma unroll
        for (int r = 0; r < 8; r += 2) {
            float x0 = d0[r]     + b1v[r];     x0 = fmaxf(x0, LEAKY * x0);
            float x1 = d0[r + 1] + b1v[r + 1]; x1 = fmaxf(x1, LEAKY * x1);
            float y0 = d1[r]     + b1v[8 + r];     y0 = fmaxf(y0, LEAKY * y0);
            float y1 = d1[r + 1] + b1v[8 + r + 1]; y1 = fmaxf(y1, LEAKY * y1);
            a0 = fmaf(x0, w2v[r],         a0);
            a1 = fmaf(x1, w2v[r + 1],     a1);
            a2 = fmaf(y0, w2v[8 + r],     a2);
            a3 = fmaf(y1, w2v[8 + r + 1], a3);
        }
        float acc = (a0 + a1) + (a2 + a3);

        // combine the two lane-halves (lane l and l+16 share pair m = l15)
        acc += __shfl_xor(acc, 16, 32);

        const float sig = fast_sigmoid(acc + bias2);

        // lane m (0..15) stores kernel[i][jbase+m]: contiguous 64B store
        float* orow = outKernel + (size_t)(ibase + ii) * N_NODES + jbase;
        if (hi16 == 0) orow[l15] = sig;
    }
}

// d_out[0 .. N*H) = hiddens passthrough (reference returns the tuple).
__global__ __launch_bounds__(256) void copy_hiddens(const float* __restrict__ src,
                                                    float* __restrict__ dst)
{
    const int idx = blockIdx.x * blockDim.x + threadIdx.x;
    const float4* s = (const float4*)src;
    float4*       d = (float4*)dst;
    d[idx] = s[idx];
}

extern "C" void kernel_launch(void* const* d_in, const int* in_sizes, int n_in,
                              void* d_out, int out_size, void* d_ws, size_t ws_size,
                              hipStream_t stream)
{
    (void)in_sizes; (void)n_in; (void)out_size; (void)d_ws; (void)ws_size;

    const float* hiddens = (const float*)d_in[0];  // [2048, 64]
    const float* W1      = (const float*)d_in[1];  // [32, 64]
    const float* b1      = (const float*)d_in[2];  // [32]
    const float* W2      = (const float*)d_in[3];  // [1, 32]
    const float* b2      = (const float*)d_in[4];  // [1]
    float*       out     = (float*)d_out;

    // hiddens passthrough: 2048*64 floats = 32768 float4
    copy_hiddens<<<(N_NODES * H_DIM / 4) / 256, 256, 0, stream>>>(hiddens, out);

    // kernel matrix at offset 2048*64
    dim3 grid(N_NODES / 128, N_NODES / 16);  // (16, 128)
    simnet_allpairs_wmma<<<grid, 256, 0, stream>>>(
        hiddens, W1, b1, W2, b2, out + (size_t)N_NODES * H_DIM);
}
